// GraphTransformer_65335042507196
// MI455X (gfx1250) — compile-verified, matrix operand order checked by reading, not run on hardware
//
#include <hip/hip_runtime.h>
#include <hip/hip_bf16.h>
#include <math.h>

// ---------------- problem constants (match reference) ----------------
#define LLAYERS 5
#define EMB     300
#define EMBP    320     // EMB padded to mult of 32
#define FFD     1200    // 4*EMB
#define FFDP    1216    // padded
#define HEADS   4
#define CCH     75
#define EDGEDIM 32
#define FEAT    256
#define NGR     2000

typedef __attribute__((ext_vector_type(16))) __bf16 bf16x16;
typedef __attribute__((ext_vector_type(8)))  float  floatx8;

// ---------------- CDNA5 async global->LDS copy helpers ----------------
__device__ inline void async_b128(unsigned lds_off, const void* gptr) {
  asm volatile("global_load_async_to_lds_b128 %0, %1, off"
               :: "v"(lds_off), "v"(gptr) : "memory");
}
__device__ inline void wait_async0() {   // drain all async copies
  asm volatile("s_wait_asynccnt 0" ::: "memory");
}
__device__ inline void wait_async4() {   // wait for all but the 4 most recent
  asm volatile("s_wait_asynccnt 4" ::: "memory");
}

// ---------------- fills ----------------
__global__ void fillf_kernel(float* p, float v, long long n) {
  long long i = (long long)blockIdx.x * blockDim.x + threadIdx.x;
  if (i < n) p[i] = v;
}
__global__ void fillbf_kernel(__bf16* p, long long n) {
  long long i = (long long)blockIdx.x * blockDim.x + threadIdx.x;
  if (i < n) p[i] = (__bf16)0.f;
}

// ---------------- weight convert + transpose + pad: W[K][N] -> Wt[N][Kp] bf16 ----------------
__global__ void cvt_w_kernel(const float* __restrict__ W, __bf16* __restrict__ Wt,
                             int K, int Nn, int Kp) {
  long long idx = (long long)blockIdx.x * blockDim.x + threadIdx.x;
  long long tot = (long long)Nn * Kp;
  if (idx >= tot) return;
  int n = (int)(idx / Kp), k = (int)(idx % Kp);
  Wt[idx] = (k < K) ? (__bf16)W[(long long)k * Nn + n] : (__bf16)0.f;
}

// ---------------- embeddings ----------------
__global__ void node_embed_kernel(const int* __restrict__ x,
                                  const float* __restrict__ emb_atom,
                                  const float* __restrict__ emb_chi,
                                  float* __restrict__ h, __bf16* __restrict__ h_bf, int N) {
  long long idx = (long long)blockIdx.x * blockDim.x + threadIdx.x;
  long long tot = (long long)N * EMB;
  if (idx >= tot) return;
  int i = (int)(idx / EMB), j = (int)(idx % EMB);
  float v = emb_atom[(long long)x[i * 2 + 0] * EMB + j] +
            emb_chi [(long long)x[i * 2 + 1] * EMB + j];
  h[idx] = v;
  h_bf[(long long)i * EMBP + j] = (__bf16)v;
}

__global__ void edge_embed_kernel(const int* __restrict__ ea,
                                  const float* __restrict__ emb_bt,
                                  const float* __restrict__ emb_bd,
                                  const float* __restrict__ W_ep,
                                  const float* __restrict__ b_ep,
                                  __bf16* __restrict__ ef_bf, int E) {
  long long idx = (long long)blockIdx.x * blockDim.x + threadIdx.x;
  long long tot = (long long)E * EDGEDIM;
  if (idx >= tot) return;
  int e = (int)(idx >> 5), j = (int)(idx & 31);
  int a0 = ea[e * 2 + 0], a1 = ea[e * 2 + 1];
  float acc = b_ep[j];
  #pragma unroll 8
  for (int t = 0; t < EDGEDIM; ++t)
    acc += (emb_bt[a0 * EDGEDIM + t] + emb_bd[a1 * EDGEDIM + t]) * W_ep[t * EDGEDIM + j];
  ef_bf[idx] = (__bf16)acc;
}

// ---------------- WMMA bf16 GEMM (double-buffered async->LDS pipeline) ----------------
// A:  [M][Kp] bf16 (K-padded with zeros), Bt: [Nn][Kp] bf16 (transposed weights, K-padded)
// Cf  (opt): f32 [M][Nn] = act(A@B + bias + res0 + res1)
// Cbf (opt): bf16 [M][Np] same values
#define BM 128
#define BN 128
#define BKT 32
#define ATILE (BM * BKT)
#define BTILE (BN * BKT)

__global__ __launch_bounds__(256)
void wmma_gemm_kernel(const __bf16* __restrict__ A, const __bf16* __restrict__ Bt,
                      const float* __restrict__ bias,
                      const float* __restrict__ res0, const float* __restrict__ res1,
                      float* __restrict__ Cf, __bf16* __restrict__ Cbf,
                      int M, int Kp, int Nn, int Np, int act) {
  __shared__ __align__(16) __bf16 As[2 * ATILE];   // [buf][m][k]
  __shared__ __align__(16) __bf16 Bs[2 * BTILE];   // [buf][n][k]

  const int tid  = threadIdx.x;
  const int wave = tid >> 5;
  const int lane = tid & 31;
  const int wm   = wave >> 1;        // 0..3 : 32-row strip
  const int wn   = wave & 1;         // 0..1 : 64-col strip
  const int m0   = blockIdx.x * BM;
  const int n0   = blockIdx.y * BN;

  floatx8 acc[2][4];
  #pragma unroll
  for (int i = 0; i < 2; ++i)
    #pragma unroll
    for (int j = 0; j < 4; ++j)
      #pragma unroll
      for (int r = 0; r < 8; ++r) acc[i][j][r] = 0.f;

  // staging assignment: 512 16-byte chunks per tile; thread t handles chunks t, t+256.
  // chunk c -> row r = c>>2, 16B-quarter q = c&3. OOB rows clamp to a valid row
  // (they only feed OOB outputs which are never stored); K needs no check (padded).
  const __bf16* aG[2]; const __bf16* bG[2];
  unsigned aL[2], bL[2];
  const unsigned asBase = (unsigned)(unsigned long long)(const void*)&As[0];
  const unsigned bsBase = (unsigned)(unsigned long long)(const void*)&Bs[0];
  #pragma unroll
  for (int t = 0; t < 2; ++t) {
    int c = tid + t * 256;
    int r = c >> 2, qd = c & 3;
    int gm = m0 + r; if (gm >= M)  gm = M - 1;
    int gn = n0 + r; if (gn >= Nn) gn = Nn - 1;
    aG[t] = A  + (long long)gm * Kp + qd * 8;
    bG[t] = Bt + (long long)gn * Kp + qd * 8;
    aL[t] = asBase + (unsigned)(r * BKT + qd * 8) * 2u;
    bL[t] = bsBase + (unsigned)(r * BKT + qd * 8) * 2u;
  }

  const int kbase = (lane < 16) ? 0 : 8;   // 16-bit fragment K-half per lane group
  const int frow  = lane & 15;
  const int nTiles = Kp / BKT;             // Kp is a multiple of BKT

  // prologue: stage tile 0 into buffer 0 (4 async copies per wave)
  #pragma unroll
  for (int t = 0; t < 2; ++t) {
    async_b128(aL[t], aG[t]);
    async_b128(bL[t], bG[t]);
  }

  for (int it = 0; it < nTiles; ++it) {
    const int buf = it & 1;
    const bool pre = (it + 1 < nTiles);
    if (pre) {
      // prefetch next tile into the other buffer (safe: that buffer was last
      // read in iteration it-1, fenced by the trailing barrier below)
      const int kt2 = (it + 1) * BKT;
      const unsigned bufOff = (unsigned)((buf ^ 1) ? 1 : 0);
      #pragma unroll
      for (int t = 0; t < 2; ++t) {
        async_b128(aL[t] + bufOff * (ATILE * 2u), aG[t] + kt2);
        async_b128(bL[t] + bufOff * (BTILE * 2u), bG[t] + kt2);
      }
      wait_async4();   // async loads complete in order: oldest 4 (tile it) done
    } else {
      wait_async0();
    }
    __syncthreads();

    const __bf16* Ab = &As[buf * ATILE];
    const __bf16* Bb = &Bs[buf * BTILE];
    bf16x16 af[2], bfr[4];
    #pragma unroll
    for (int i = 0; i < 2; ++i) {
      int row = wm * 32 + i * 16 + frow;
      union { bf16x16 v; uint4 u[2]; } t;
      t.u[0] = *(const uint4*)(&Ab[row * BKT + kbase]);
      t.u[1] = *(const uint4*)(&Ab[row * BKT + kbase + 16]);
      af[i] = t.v;
    }
    #pragma unroll
    for (int j = 0; j < 4; ++j) {
      int col = wn * 64 + j * 16 + frow;
      union { bf16x16 v; uint4 u[2]; } t;
      t.u[0] = *(const uint4*)(&Bb[col * BKT + kbase]);
      t.u[1] = *(const uint4*)(&Bb[col * BKT + kbase + 16]);
      bfr[j] = t.v;
    }
    #pragma unroll
    for (int i = 0; i < 2; ++i)
      #pragma unroll
      for (int j = 0; j < 4; ++j)
        acc[i][j] = __builtin_amdgcn_wmma_f32_16x16x32_bf16(
            false, af[i], false, bfr[j], (short)0, acc[i][j], false, false);
    __syncthreads();
  }

  // epilogue: C/D f32 layout: vgpr r, lanes 0-15 -> M=r, lanes 16-31 -> M=8+r
  const int nlane = lane & 15;
  const int mhalf = (lane >> 4) * 8;
  #pragma unroll
  for (int i = 0; i < 2; ++i)
    #pragma unroll
    for (int j = 0; j < 4; ++j) {
      int gn = n0 + wn * 64 + j * 16 + nlane;
      if (gn >= Nn) continue;
      float bval = bias ? bias[gn] : 0.f;
      #pragma unroll
      for (int r = 0; r < 8; ++r) {
        int gm = m0 + wm * 32 + i * 16 + mhalf + r;
        if (gm < M) {
          long long off = (long long)gm * Nn + gn;
          float vv = acc[i][j][r] + bval;
          if (res0) vv += res0[off];
          if (res1) vv += res1[off];
          if (act) vv = fmaxf(vv, 0.f);
          if (Cf)  Cf[off] = vv;
          if (Cbf) Cbf[(long long)gm * Np + gn] = (__bf16)vv;
        }
      }
    }
}

// ---------------- attention ----------------
__device__ inline void atomicMaxF(float* addr, float val) {
  if (val >= 0.f) atomicMax((int*)addr, __float_as_int(val));
  else            atomicMin((unsigned int*)addr, (unsigned int)__float_as_int(val));
}

__global__ void attn_alpha_kernel(const float* __restrict__ q, const float* __restrict__ k,
                                  const float* __restrict__ e,
                                  const int* __restrict__ src, const int* __restrict__ dst,
                                  float* __restrict__ alpha, int E, float scale) {
  int idx = blockIdx.x * blockDim.x + threadIdx.x;
  if (idx >= E * HEADS) return;
  int ed = idx / HEADS, hh = idx % HEADS;
  int s = src[ed], d = dst[ed];
  const float* qp = q + (long long)d  * EMB + hh * CCH;
  const float* kp = k + (long long)s  * EMB + hh * CCH;
  const float* ep = e + (long long)ed * EMB + hh * CCH;
  float acc = 0.f;
  for (int c = 0; c < CCH; ++c) acc += qp[c] * (kp[c] + ep[c]);
  alpha[idx] = acc * scale;
}

__global__ void attn_max_kernel(const float* __restrict__ alpha, const int* __restrict__ dst,
                                float* __restrict__ amax, int E) {
  int idx = blockIdx.x * blockDim.x + threadIdx.x;
  if (idx >= E * HEADS) return;
  int ed = idx / HEADS, hh = idx % HEADS;
  atomicMaxF(&amax[dst[ed] * HEADS + hh], alpha[idx]);
}

__global__ void attn_expsum_kernel(const float* __restrict__ alpha, const int* __restrict__ dst,
                                   const float* __restrict__ amax,
                                   float* __restrict__ aexp, float* __restrict__ asum, int E) {
  int idx = blockIdx.x * blockDim.x + threadIdx.x;
  if (idx >= E * HEADS) return;
  int ed = idx / HEADS, hh = idx % HEADS;
  int d = dst[ed];
  float ex = __expf(alpha[idx] - amax[d * HEADS + hh]);
  aexp[idx] = ex;
  atomicAdd(&asum[d * HEADS + hh], ex);
}

__global__ void attn_msg_kernel(const float* __restrict__ v, const float* __restrict__ e,
                                const int* __restrict__ src, const int* __restrict__ dst,
                                const float* __restrict__ aexp, const float* __restrict__ asum,
                                float* __restrict__ agg, int E) {
  int idx = blockIdx.x * blockDim.x + threadIdx.x;
  if (idx >= E * HEADS) return;
  int ed = idx / HEADS, hh = idx % HEADS;
  int s = src[ed], d = dst[ed];
  float w = aexp[idx] / (asum[d * HEADS + hh] + 1e-16f);
  const float* vp = v + (long long)s  * EMB + hh * CCH;
  const float* ep = e + (long long)ed * EMB + hh * CCH;
  float* ap = agg + (long long)d * EMB + hh * CCH;
  for (int c = 0; c < CCH; ++c) atomicAdd(&ap[c], w * (vp[c] + ep[c]));
}

// ---------------- layernorm: one wave per row, dual f32/bf16 output ----------------
__global__ __launch_bounds__(128)
void layernorm_kernel(const float* __restrict__ in, float* __restrict__ out,
                      __bf16* __restrict__ out_bf,
                      const float* __restrict__ gamma, const float* __restrict__ beta,
                      int rows) {
  int row = blockIdx.x * 4 + (threadIdx.x >> 5);
  int lane = threadIdx.x & 31;
  if (row >= rows) return;
  const float* p = in + (long long)row * EMB;
  float s = 0.f, s2 = 0.f;
  for (int j = lane; j < EMB; j += 32) { float x = p[j]; s += x; s2 += x * x; }
  #pragma unroll
  for (int o = 16; o > 0; o >>= 1) {
    s  += __shfl_xor(s,  o, 32);
    s2 += __shfl_xor(s2, o, 32);
  }
  float mean = s * (1.f / EMB);
  float var  = s2 * (1.f / EMB) - mean * mean;
  float inv  = rsqrtf(var + 1e-5f);
  float* qo = out + (long long)row * EMB;
  __bf16* qb = out_bf + (long long)row * EMBP;
  for (int j = lane; j < EMB; j += 32) {
    float y = (p[j] - mean) * inv * gamma[j] + beta[j];
    qo[j] = y;
    qb[j] = (__bf16)y;
  }
}

// ---------------- pooling ----------------
__global__ void pool_count_kernel(const int* __restrict__ batch, float* __restrict__ cnt, int N) {
  int i = blockIdx.x * blockDim.x + threadIdx.x;
  if (i < N) atomicAdd(&cnt[batch[i]], 1.f);
}
__global__ void pool_sum_kernel(const float* __restrict__ h, const int* __restrict__ batch,
                                float* __restrict__ gsum, int N) {
  long long idx = (long long)blockIdx.x * blockDim.x + threadIdx.x;
  long long tot = (long long)N * EMB;
  if (idx >= tot) return;
  int i = (int)(idx / EMB), j = (int)(idx % EMB);
  atomicAdd(&gsum[(long long)batch[i] * EMB + j], h[idx]);
}
__global__ void pool_div_kernel(const float* __restrict__ gsum, const float* __restrict__ cnt,
                                __bf16* __restrict__ gpool_bf) {
  long long idx = (long long)blockIdx.x * blockDim.x + threadIdx.x;
  if (idx >= (long long)NGR * EMB) return;
  int g = (int)(idx / EMB), j = (int)(idx % EMB);
  gpool_bf[(long long)g * EMBP + j] = (__bf16)(gsum[idx] / fmaxf(cnt[g], 1.f));
}

// ---------------- host-side helpers ----------------
static inline void launch_gemm(hipStream_t s, const __bf16* A, const __bf16* Bt,
                               const float* bias, const float* r0, const float* r1,
                               float* Cf, __bf16* Cbf,
                               int M, int Kp, int Nn, int Np, int act) {
  dim3 grid((M + BM - 1) / BM, (Nn + BN - 1) / BN);
  wmma_gemm_kernel<<<grid, 256, 0, s>>>(A, Bt, bias, r0, r1, Cf, Cbf, M, Kp, Nn, Np, act);
}
static inline void launch_fillf(hipStream_t s, float* p, float v, long long n) {
  fillf_kernel<<<(unsigned)((n + 255) / 256), 256, 0, s>>>(p, v, n);
}
static inline void launch_fillbf(hipStream_t s, __bf16* p, long long n) {
  fillbf_kernel<<<(unsigned)((n + 255) / 256), 256, 0, s>>>(p, n);
}

extern "C" void kernel_launch(void* const* d_in, const int* in_sizes, int n_in,
                              void* d_out, int out_size, void* d_ws, size_t ws_size,
                              hipStream_t stream) {
  // -------- inputs (setup_inputs dict order; params in insertion order) --------
  const int* x          = (const int*)d_in[0];
  const int* edge_index = (const int*)d_in[1];
  const int* edge_attr  = (const int*)d_in[2];
  const int* batch      = (const int*)d_in[3];
  const float* emb_atom = (const float*)d_in[4];
  const float* emb_chi  = (const float*)d_in[5];
  const float* emb_bt   = (const float*)d_in[6];
  const float* emb_bd   = (const float*)d_in[7];
  const float* W_ep     = (const float*)d_in[8];
  const float* b_ep     = (const float*)d_in[9];
  const float* Wq    = (const float*)d_in[10];
  const float* bq    = (const float*)d_in[11];
  const float* Wk    = (const float*)d_in[12];
  const float* bk    = (const float*)d_in[13];
  const float* Wv    = (const float*)d_in[14];
  const float* bv    = (const float*)d_in[15];
  const float* We    = (const float*)d_in[16];
  const float* Wskip = (const float*)d_in[17];
  const float* bskip = (const float*)d_in[18];
  const float* g1    = (const float*)d_in[19];
  const float* be1   = (const float*)d_in[20];
  const float* g2    = (const float*)d_in[21];
  const float* be2   = (const float*)d_in[22];
  const float* Wf1   = (const float*)d_in[23];
  const float* bf1   = (const float*)d_in[24];
  const float* Wf2   = (const float*)d_in[25];
  const float* bf2   = (const float*)d_in[26];
  const float* Wfeat = (const float*)d_in[27];
  const float* bfeat = (const float*)d_in[28];
  const float* Wo1   = (const float*)d_in[29];
  const float* bo1   = (const float*)d_in[30];
  const float* Wo2   = (const float*)d_in[31];
  const float* bo2   = (const float*)d_in[32];

  const int N = in_sizes[0] / 2;
  const int E = in_sizes[1] / 2;
  const int* src = edge_index;       // row 0
  const int* dst = edge_index + E;   // row 1

  // -------- workspace layout (byte allocator, 256B aligned) --------
  char* base = (char*)d_ws;
  size_t off = 0;
  auto allocB = [&](long long bytes) {
    char* p = base + off;
    off += (size_t)((bytes + 255) & ~255LL);
    return (void*)p;
  };
  // f32 buffers
  float* h     = (float*)allocB((long long)N * EMB * 4);
  float* tmp   = (float*)allocB((long long)N * EMB * 4);
  float* q     = (float*)allocB((long long)N * EMB * 4);
  float* k     = (float*)allocB((long long)N * EMB * 4);
  float* v     = (float*)allocB((long long)N * EMB * 4);
  float* agg   = (float*)allocB((long long)N * EMB * 4);
  float* ee    = (float*)allocB((long long)E * EMB * 4);
  float* alpha = (float*)allocB((long long)E * HEADS * 4);
  float* aexp  = (float*)allocB((long long)E * HEADS * 4);
  float* amax  = (float*)allocB((long long)N * HEADS * 4);
  float* asum  = (float*)allocB((long long)N * HEADS * 4);
  float* cnt   = (float*)allocB((long long)NGR * 4);
  float* gsum  = (float*)allocB((long long)NGR * EMB * 4);
  // bf16 activation buffers (K-padded strides)
  __bf16* h_bf     = (__bf16*)allocB((long long)N * EMBP * 2);
  __bf16* ffmid_bf = (__bf16*)allocB((long long)N * FFDP * 2);
  __bf16* ef_bf    = (__bf16*)allocB((long long)E * EDGEDIM * 2);
  __bf16* gpool_bf = (__bf16*)allocB((long long)NGR * EMBP * 2);
  __bf16* outg_bf  = (__bf16*)allocB((long long)NGR * FEAT * 2);
  __bf16* go1_bf   = (__bf16*)allocB((long long)NGR * FEAT * 2);
  // bf16 transposed, K-padded weights
  __bf16* Wq_t    = (__bf16*)allocB((long long)LLAYERS * EMB * EMBP * 2);
  __bf16* Wk_t    = (__bf16*)allocB((long long)LLAYERS * EMB * EMBP * 2);
  __bf16* Wv_t    = (__bf16*)allocB((long long)LLAYERS * EMB * EMBP * 2);
  __bf16* Ws_t    = (__bf16*)allocB((long long)LLAYERS * EMB * EMBP * 2);
  __bf16* We_t    = (__bf16*)allocB((long long)LLAYERS * EMB * EDGEDIM * 2);
  __bf16* Wf1_t   = (__bf16*)allocB((long long)LLAYERS * FFD * EMBP * 2);
  __bf16* Wf2_t   = (__bf16*)allocB((long long)LLAYERS * EMB * FFDP * 2);
  __bf16* Wfeat_t = (__bf16*)allocB((long long)FEAT * EMBP * 2);
  __bf16* Wo1_t   = (__bf16*)allocB((long long)FEAT * FEAT * 2);
  __bf16* Wo2_t   = (__bf16*)allocB((long long)(FEAT / 2) * FEAT * 2);
  (void)ws_size;

  float* out_g   = (float*)d_out;                          // [NGR, FEAT]
  float* out_mlp = (float*)d_out + (long long)NGR * FEAT;  // [NGR, FEAT/2]

  const float scale = 1.f / sqrtf((float)CCH);
  const long long NE = (long long)N * EMB;
  const int EH = E * HEADS;

  // -------- zero-fill padded bf16 activation buffers (pads must be 0) --------
  launch_fillbf(stream, h_bf, (long long)N * EMBP);
  launch_fillbf(stream, ffmid_bf, (long long)N * FFDP);
  launch_fillbf(stream, gpool_bf, (long long)NGR * EMBP);

  // -------- convert + transpose + pad weights to bf16 --------
  auto cvtw = [&](const float* Ws, __bf16* Wd, int K, int Nn, int Kp) {
    long long n = (long long)Nn * Kp;
    cvt_w_kernel<<<(unsigned)((n + 255) / 256), 256, 0, stream>>>(Ws, Wd, K, Nn, Kp);
  };
  for (int l = 0; l < LLAYERS; ++l) {
    cvtw(Wq + (long long)l * EMB * EMB, Wq_t + (long long)l * EMB * EMBP, EMB, EMB, EMBP);
    cvtw(Wk + (long long)l * EMB * EMB, Wk_t + (long long)l * EMB * EMBP, EMB, EMB, EMBP);
    cvtw(Wv + (long long)l * EMB * EMB, Wv_t + (long long)l * EMB * EMBP, EMB, EMB, EMBP);
    cvtw(Wskip + (long long)l * EMB * EMB, Ws_t + (long long)l * EMB * EMBP, EMB, EMB, EMBP);
    cvtw(We + (long long)l * EDGEDIM * EMB, We_t + (long long)l * EMB * EDGEDIM, EDGEDIM, EMB, EDGEDIM);
    cvtw(Wf1 + (long long)l * EMB * FFD, Wf1_t + (long long)l * FFD * EMBP, EMB, FFD, EMBP);
    cvtw(Wf2 + (long long)l * FFD * EMB, Wf2_t + (long long)l * EMB * FFDP, FFD, EMB, FFDP);
  }
  cvtw(Wfeat, Wfeat_t, EMB, FEAT, EMBP);
  cvtw(Wo1, Wo1_t, FEAT, FEAT, FEAT);
  cvtw(Wo2, Wo2_t, FEAT, FEAT / 2, FEAT);

  // -------- embeddings --------
  node_embed_kernel<<<(unsigned)((NE + 255) / 256), 256, 0, stream>>>(x, emb_atom, emb_chi, h, h_bf, N);
  edge_embed_kernel<<<(unsigned)(((long long)E * EDGEDIM + 255) / 256), 256, 0, stream>>>(
      edge_attr, emb_bt, emb_bd, W_ep, b_ep, ef_bf, E);

  // -------- layers --------
  for (int l = 0; l < LLAYERS; ++l) {
    const __bf16* Wq_l = Wq_t + (long long)l * EMB * EMBP;
    const __bf16* Wk_l = Wk_t + (long long)l * EMB * EMBP;
    const __bf16* Wv_l = Wv_t + (long long)l * EMB * EMBP;
    const __bf16* We_l = We_t + (long long)l * EMB * EDGEDIM;
    const __bf16* Ws_l = Ws_t + (long long)l * EMB * EMBP;
    const __bf16* Wf1_l = Wf1_t + (long long)l * FFD * EMBP;
    const __bf16* Wf2_l = Wf2_t + (long long)l * EMB * FFDP;

    launch_gemm(stream, h_bf, Wq_l, bq + l * EMB, nullptr, nullptr, q, nullptr, N, EMBP, EMB, 0, 0);
    launch_gemm(stream, h_bf, Wk_l, bk + l * EMB, nullptr, nullptr, k, nullptr, N, EMBP, EMB, 0, 0);
    launch_gemm(stream, h_bf, Wv_l, bv + l * EMB, nullptr, nullptr, v, nullptr, N, EMBP, EMB, 0, 0);
    launch_gemm(stream, ef_bf, We_l, nullptr, nullptr, nullptr, ee, nullptr, E, EDGEDIM, EMB, 0, 0);

    launch_fillf(stream, amax, -INFINITY, (long long)N * HEADS);
    launch_fillf(stream, asum, 0.f, (long long)N * HEADS);
    launch_fillf(stream, agg, 0.f, NE);

    attn_alpha_kernel<<<(EH + 255) / 256, 256, 0, stream>>>(q, k, ee, src, dst, alpha, E, scale);
    attn_max_kernel<<<(EH + 255) / 256, 256, 0, stream>>>(alpha, dst, amax, E);
    attn_expsum_kernel<<<(EH + 255) / 256, 256, 0, stream>>>(alpha, dst, amax, aexp, asum, E);
    attn_msg_kernel<<<(EH + 255) / 256, 256, 0, stream>>>(v, ee, src, dst, aexp, asum, agg, E);

    // tmp = h@Wskip + bskip + agg + h ; h = LN(tmp)
    launch_gemm(stream, h_bf, Ws_l, bskip + l * EMB, agg, h, tmp, nullptr, N, EMBP, EMB, 0, 0);
    layernorm_kernel<<<(N + 3) / 4, 128, 0, stream>>>(tmp, h, h_bf, g1 + l * EMB, be1 + l * EMB, N);

    // FFN: ffmid = relu(h@Wf1+b) [bf16 only]; tmp = ffmid@Wf2 + b + h; h = LN(tmp)
    launch_gemm(stream, h_bf, Wf1_l, bf1 + (long long)l * FFD, nullptr, nullptr,
                nullptr, ffmid_bf, N, EMBP, FFD, FFDP, 1);
    launch_gemm(stream, ffmid_bf, Wf2_l, bf2 + l * EMB, h, nullptr, tmp, nullptr, N, FFDP, EMB, 0, 0);
    layernorm_kernel<<<(N + 3) / 4, 128, 0, stream>>>(tmp, h, h_bf, g2 + l * EMB, be2 + l * EMB, N);
  }

  // -------- global mean pool --------
  launch_fillf(stream, cnt, 0.f, NGR);
  launch_fillf(stream, gsum, 0.f, (long long)NGR * EMB);
  pool_count_kernel<<<(N + 255) / 256, 256, 0, stream>>>(batch, cnt, N);
  pool_sum_kernel<<<(unsigned)((NE + 255) / 256), 256, 0, stream>>>(h, batch, gsum, N);
  pool_div_kernel<<<(unsigned)(((long long)NGR * EMB + 255) / 256), 256, 0, stream>>>(gsum, cnt, gpool_bf);

  // -------- output head --------
  launch_gemm(stream, gpool_bf, Wfeat_t, bfeat, nullptr, nullptr, out_g, outg_bf,
              NGR, EMBP, FEAT, FEAT, 0);                                            // g (output 1)
  launch_gemm(stream, outg_bf, Wo1_t, bo1, nullptr, nullptr, nullptr, go1_bf,
              NGR, FEAT, FEAT, FEAT, 1);                                            // relu hidden
  launch_gemm(stream, go1_bf, Wo2_t, bo2, nullptr, nullptr, out_mlp, nullptr,
              NGR, FEAT, FEAT / 2, 0, 0);                                           // out (output 2)
  (void)out_size; (void)n_in;
}
